// MyBarDecoder_75239237091461
// MI455X (gfx1250) — compile-verified, matrix operand order checked by reading, not run on hardware
//
#include <hip/hip_runtime.h>
#include <hip/hip_bf16.h>
#include <math.h>

typedef __bf16 bf16_t;
typedef __attribute__((ext_vector_type(16))) __bf16 v16bf;
typedef __attribute__((ext_vector_type(8)))  float  v8f;

// ---------------- problem constants ----------------
#define kB       8192
#define kSeq     64
#define kZD      256
#define kH       64
#define kNN      128
#define kEncMid  160
#define kDecMid  96
#define kG4      256           // 4*H
#define kRowsPerBlock 128      // 8 waves * 16 rows
#define kBlocks  (kB / kRowsPerBlock)

// ---------------- workspace layout (bf16 element offsets) ----------------
#define WS_WE1   0
#define WS_WE2   (WS_WE1  + kEncMid*kZD)     // 40960
#define WS_WIH0  (WS_WE2  + kH*kEncMid)      // 51200
#define WS_WHH0  (WS_WIH0 + kG4*kH)
#define WS_WIH1  (WS_WHH0 + kG4*kH)
#define WS_WHH1  (WS_WIH1 + kG4*kH)
#define WS_WD1   (WS_WHH1 + kG4*kH)
#define WS_WD2   (WS_WD1  + kDecMid*kH)
#define WS_BF16_TOTAL (WS_WD2 + kNN*kDecMid) // 135168 bf16 elements

// ---------------- shared memory layout (bytes) ----------------
#define SM_WIH0  0
#define SM_WHH0  (SM_WIH0 + kG4*kH*2)
#define SM_WIH1  (SM_WHH0 + kG4*kH*2)
#define SM_WHH1  (SM_WIH1 + kG4*kH*2)
#define SM_WD1   (SM_WHH1 + kG4*kH*2)        // 131072
#define SM_WD2   (SM_WD1  + kDecMid*kH*2)    // 143360
#define SM_H0    (SM_WD2  + kNN*kDecMid*2)   // 167936
#define SM_H1    (SM_H0   + kRowsPerBlock*kH*2)
#define SM_A1    (SM_H1   + kRowsPerBlock*kH*2)
#define SM_BIAS  (SM_A1   + kRowsPerBlock*kDecMid*2)  // 225280 (f32 region)
#define BO_G0    0
#define BO_G1    256
#define BO_D1    512
#define BO_D2    608
#define BO_E1    736
#define BO_E2    896
#define SM_TOTAL (SM_BIAS + 960*4)           // 229120 bytes < 320KB WGP LDS
// encoder-phase aliases (overwritten by LSTM weights after a barrier)
#define SM_WE1   0
#define SM_WE2   (SM_WE1 + kEncMid*kZD*2)    // 81920
#define SM_HID   (SM_WE2 + kH*kEncMid*2)     // 102400..143360

// ---------------- helpers ----------------
union FragU { v16bf v; uint4 u[2]; };

__device__ __forceinline__ v8f vzero8() {
  v8f a;
#pragma unroll
  for (int i = 0; i < 8; ++i) a[i] = 0.0f;
  return a;
}

__device__ __forceinline__ float sigm(float x) { return 1.0f / (1.0f + __expf(-x)); }

__device__ __forceinline__ unsigned wang(unsigned s) {
  s = (s ^ 61u) ^ (s >> 16); s *= 9u; s ^= s >> 4; s *= 0x27d4eb2du; s ^= s >> 15;
  return s;
}

// WMMA D = A(16x32 bf16) * B(32x16 bf16) + C(16x16 f32)
__device__ __forceinline__ v8f wmma_bf16(v16bf a, v16bf b, v8f c) {
  return __builtin_amdgcn_wmma_f32_16x16x32_bf16(false, a, false, b, (short)0, c, false, false);
}

// Load a 16x32 A-fragment (or 32x16 B-fragment; same lane pattern with lane=N)
// from a row-major bf16 matrix. ISA 7.12.2 16-bit layout: lane group g=lane>>4
// holds K = k0+g*8..+7 (16B) and K = k0+16+g*8..+7 (16B) -> two b128 loads.
__device__ __forceinline__ v16bf lds_frag(const bf16_t* base, int row0, int k0,
                                          int stride, int lane) {
  const int g = lane >> 4, r = lane & 15;
  const bf16_t* p = base + (size_t)(row0 + r) * stride + k0 + g * 8;
  FragU f;
  f.u[0] = *(const uint4*)(p);
  f.u[1] = *(const uint4*)(p + 16);
  return f.v;
}

// Same fragment pattern from a row-major f32 matrix in global memory (converts to bf16).
__device__ __forceinline__ v16bf glb_fragf32(const float* base, long row0, int k0,
                                             int stride, int lane) {
  const int g = lane >> 4, r = lane & 15;
  const float* p = base + (row0 + r) * (long)stride + k0 + g * 8;
  const float4* q = (const float4*)p;
  float4 f0 = q[0], f1 = q[1], f2 = q[4], f3 = q[5];
  float tmp[16] = {f0.x, f0.y, f0.z, f0.w, f1.x, f1.y, f1.z, f1.w,
                   f2.x, f2.y, f2.z, f2.w, f3.x, f3.y, f3.z, f3.w};
  v16bf v;
#pragma unroll
  for (int e = 0; e < 16; ++e) v[e] = (bf16_t)tmp[e];
  return v;
}

// Scatter a C/D tile (lane=N, vgpr r -> row r or r+8) into row-major bf16 LDS.
__device__ __forceinline__ void storeC(bf16_t* base, int row0, int n0, int stride,
                                       int lane, v8f acc) {
  const int n = lane & 15, mh = (lane >> 4) * 8;
#pragma unroll
  for (int r = 0; r < 8; ++r)
    base[(size_t)(row0 + mh + r) * stride + n0 + n] = (bf16_t)acc[r];
}

__device__ __forceinline__ void cp16(void* dst, const void* src, int bytes,
                                     int tid, int nthr) {
  uint4* d = (uint4*)dst;
  const uint4* s = (const uint4*)src;
  const int n = bytes >> 4;
  for (int i = tid; i < n; i += nthr) d[i] = s[i];
}

// LSTM elementwise update for one gate-quadrant-aligned tile group.
__device__ __forceinline__ v8f lstm_pointwise(v8f ai, v8f af, v8f ag, v8f ao,
                                              v8f& cst) {
  v8f hn;
#pragma unroll
  for (int r = 0; r < 8; ++r) {
    const float iv = sigm(ai[r]);
    const float fv = sigm(af[r]);
    const float gv = tanhf(ag[r]);
    const float ov = sigm(ao[r]);
    const float cv = fv * cst[r] + iv * gv;
    cst[r] = cv;
    hn[r] = ov * tanhf(cv);
  }
  return hn;
}

// Layer 0: x@Wih.T + bias is loop-invariant (x = encoded) and pre-folded into
// gx[q][p]; only the recurrent h@Whh.T term is computed each step (32 WMMAs).
__device__ __forceinline__ void lstm_layer0(const v8f (&gx)[4][4], bf16_t* s_h,
                                            const bf16_t* s_whh, v8f (&c)[4],
                                            int wloc, int lane) {
  v16bf ha0 = lds_frag(s_h, wloc, 0, kH, lane);
  v16bf ha1 = lds_frag(s_h, wloc, 32, kH, lane);
#pragma unroll
  for (int p = 0; p < 4; ++p) {
    v8f ai = gx[0][p], af = gx[1][p], ag = gx[2][p], ao = gx[3][p];
    ai = wmma_bf16(ha0, lds_frag(s_whh, (p + 0) * 16, 0, kH, lane), ai);
    ai = wmma_bf16(ha1, lds_frag(s_whh, (p + 0) * 16, 32, kH, lane), ai);
    af = wmma_bf16(ha0, lds_frag(s_whh, (p + 4) * 16, 0, kH, lane), af);
    af = wmma_bf16(ha1, lds_frag(s_whh, (p + 4) * 16, 32, kH, lane), af);
    ag = wmma_bf16(ha0, lds_frag(s_whh, (p + 8) * 16, 0, kH, lane), ag);
    ag = wmma_bf16(ha1, lds_frag(s_whh, (p + 8) * 16, 32, kH, lane), ag);
    ao = wmma_bf16(ha0, lds_frag(s_whh, (p + 12) * 16, 0, kH, lane), ao);
    ao = wmma_bf16(ha1, lds_frag(s_whh, (p + 12) * 16, 32, kH, lane), ao);
    v8f hn = lstm_pointwise(ai, af, ag, ao, c[p]);
    storeC(s_h, wloc, p * 16, kH, lane, hn);
  }
}

// Layer 1: dynamic input x (= h0 of this step) -> full 64-WMMA gate matmul.
__device__ __forceinline__ void lstm_layer1(v16bf xa0, v16bf xa1, bf16_t* s_h,
                                            const bf16_t* s_wih, const bf16_t* s_whh,
                                            const float* bias, v8f (&c)[4],
                                            int wloc, int lane) {
  const int ln = lane & 15;
  v16bf ha0 = lds_frag(s_h, wloc, 0, kH, lane);
  v16bf ha1 = lds_frag(s_h, wloc, 32, kH, lane);
#pragma unroll
  for (int p = 0; p < 4; ++p) {
    v8f ai = vzero8(), af = vzero8(), ag = vzero8(), ao = vzero8();
    ai = wmma_bf16(xa0, lds_frag(s_wih, (p + 0) * 16, 0, kH, lane), ai);
    ai = wmma_bf16(xa1, lds_frag(s_wih, (p + 0) * 16, 32, kH, lane), ai);
    ai = wmma_bf16(ha0, lds_frag(s_whh, (p + 0) * 16, 0, kH, lane), ai);
    ai = wmma_bf16(ha1, lds_frag(s_whh, (p + 0) * 16, 32, kH, lane), ai);
    af = wmma_bf16(xa0, lds_frag(s_wih, (p + 4) * 16, 0, kH, lane), af);
    af = wmma_bf16(xa1, lds_frag(s_wih, (p + 4) * 16, 32, kH, lane), af);
    af = wmma_bf16(ha0, lds_frag(s_whh, (p + 4) * 16, 0, kH, lane), af);
    af = wmma_bf16(ha1, lds_frag(s_whh, (p + 4) * 16, 32, kH, lane), af);
    ag = wmma_bf16(xa0, lds_frag(s_wih, (p + 8) * 16, 0, kH, lane), ag);
    ag = wmma_bf16(xa1, lds_frag(s_wih, (p + 8) * 16, 32, kH, lane), ag);
    ag = wmma_bf16(ha0, lds_frag(s_whh, (p + 8) * 16, 0, kH, lane), ag);
    ag = wmma_bf16(ha1, lds_frag(s_whh, (p + 8) * 16, 32, kH, lane), ag);
    ao = wmma_bf16(xa0, lds_frag(s_wih, (p + 12) * 16, 0, kH, lane), ao);
    ao = wmma_bf16(xa1, lds_frag(s_wih, (p + 12) * 16, 32, kH, lane), ao);
    ao = wmma_bf16(ha0, lds_frag(s_whh, (p + 12) * 16, 0, kH, lane), ao);
    ao = wmma_bf16(ha1, lds_frag(s_whh, (p + 12) * 16, 32, kH, lane), ao);
    const float bi = bias[(p + 0) * 16 + ln], bff = bias[(p + 4) * 16 + ln];
    const float bgg = bias[(p + 8) * 16 + ln], bo = bias[(p + 12) * 16 + ln];
#pragma unroll
    for (int r = 0; r < 8; ++r) {
      ai[r] += bi; af[r] += bff; ag[r] += bgg; ao[r] += bo;
    }
    v8f hn = lstm_pointwise(ai, af, ag, ao, c[p]);
    storeC(s_h, wloc, p * 16, kH, lane, hn);
  }
}

// ---------------- prep: f32 weights -> bf16 workspace; fold gate biases ----------------
__global__ void bar_decoder_prep(const float* __restrict__ we1, const float* __restrict__ we2,
                                 const float* __restrict__ wih0, const float* __restrict__ whh0,
                                 const float* __restrict__ wih1, const float* __restrict__ whh1,
                                 const float* __restrict__ wd1, const float* __restrict__ wd2,
                                 const float* __restrict__ bih0, const float* __restrict__ bhh0,
                                 const float* __restrict__ bih1, const float* __restrict__ bhh1,
                                 bf16_t* __restrict__ wsb, float* __restrict__ bg) {
  const int i = blockIdx.x * blockDim.x + threadIdx.x;
  if (i < kEncMid * kZD) wsb[WS_WE1 + i] = (bf16_t)we1[i];
  if (i < kH * kEncMid) wsb[WS_WE2 + i] = (bf16_t)we2[i];
  if (i < kG4 * kH) {
    wsb[WS_WIH0 + i] = (bf16_t)wih0[i];
    wsb[WS_WHH0 + i] = (bf16_t)whh0[i];
    wsb[WS_WIH1 + i] = (bf16_t)wih1[i];
    wsb[WS_WHH1 + i] = (bf16_t)whh1[i];
  }
  if (i < kDecMid * kH) wsb[WS_WD1 + i] = (bf16_t)wd1[i];
  if (i < kNN * kDecMid) wsb[WS_WD2 + i] = (bf16_t)wd2[i];
  if (i < kG4) {
    bg[i] = bih0[i] + bhh0[i];
    bg[kG4 + i] = bih1[i] + bhh1[i];
  }
}

// ---------------- main fused kernel ----------------
__global__ void __launch_bounds__(256)
bar_decoder_main(const float* __restrict__ z,
                 const float* __restrict__ be1, const float* __restrict__ be2,
                 const float* __restrict__ bd1, const float* __restrict__ bd2,
                 const bf16_t* __restrict__ wsb, const float* __restrict__ bg,
                 float* __restrict__ out_w, float* __restrict__ out_s) {
  extern __shared__ char smem[];
  const int tid = threadIdx.x;
  const int lane = tid & 31, w = tid >> 5;
  const int wloc = w * 16, ln = lane & 15, mh = (lane >> 4) * 8;
  const long grow = (long)blockIdx.x * kRowsPerBlock + wloc;

  bf16_t* s_h0 = (bf16_t*)(smem + SM_H0);
  bf16_t* s_h1 = (bf16_t*)(smem + SM_H1);
  bf16_t* s_a1 = (bf16_t*)(smem + SM_A1);
  float* s_bias = (float*)(smem + SM_BIAS);

  // ---- stage encoder weights (bf16) and all biases into LDS ----
  cp16(smem + SM_WE1, wsb + WS_WE1, kEncMid * kZD * 2, tid, 256);
  cp16(smem + SM_WE2, wsb + WS_WE2, kH * kEncMid * 2, tid, 256);
  for (int i = tid; i < 512; i += 256) s_bias[BO_G0 + i] = bg[i];
  if (tid < kDecMid) s_bias[BO_D1 + tid] = bd1[tid];
  if (tid < kNN) s_bias[BO_D2 + tid] = bd2[tid];
  if (tid < kEncMid) s_bias[BO_E1 + tid] = be1[tid];
  if (tid < kH) s_bias[BO_E2 + tid] = be2[tid];
  __syncthreads();

  // ---- encoder: encoded = relu(z@We1.T+b1)@We2.T+b2 (once per row) ----
  {
    bf16_t* s_we1 = (bf16_t*)(smem + SM_WE1);
    bf16_t* s_we2 = (bf16_t*)(smem + SM_WE2);
    bf16_t* s_hid = (bf16_t*)(smem + SM_HID);
    v16bf za[8];
#pragma unroll
    for (int ks = 0; ks < 8; ++ks) za[ks] = glb_fragf32(z, grow, ks * 32, kZD, lane);
#pragma unroll
    for (int tn = 0; tn < 10; ++tn) {  // ENC_MID = 160 -> 10 N-tiles
      v8f a = vzero8();
#pragma unroll
      for (int ks = 0; ks < 8; ++ks)
        a = wmma_bf16(za[ks], lds_frag(s_we1, tn * 16, ks * 32, kZD, lane), a);
      const float bb = s_bias[BO_E1 + tn * 16 + ln];
      v8f rl;
#pragma unroll
      for (int r = 0; r < 8; ++r) rl[r] = fmaxf(a[r] + bb, 0.0f);
      storeC(s_hid, wloc, tn * 16, kEncMid, lane, rl);
    }
    v16bf hf[5];
#pragma unroll
    for (int ks = 0; ks < 5; ++ks) hf[ks] = lds_frag(s_hid, wloc, ks * 32, kEncMid, lane);
#pragma unroll
    for (int tn = 0; tn < 4; ++tn) {  // H = 64 -> 4 N-tiles
      v8f a = vzero8();
#pragma unroll
      for (int ks = 0; ks < 5; ++ks)
        a = wmma_bf16(hf[ks], lds_frag(s_we2, tn * 16, ks * 32, kEncMid, lane), a);
      const float bb = s_bias[BO_E2 + tn * 16 + ln];
#pragma unroll
      for (int r = 0; r < 8; ++r) a[r] += bb;
      storeC(s_h0, wloc, tn * 16, kH, lane, a);  // stage encoded in h0 region
    }
  }
  // x (= encoded) A-fragments: constant across all 64 steps.
  v16bf xa0 = lds_frag(s_h0, wloc, 0, kH, lane);
  v16bf xa1 = lds_frag(s_h0, wloc, 32, kH, lane);
  __syncthreads();

  // ---- zero h state, stage LSTM+decoder weights (overwrites encoder LDS) ----
  {
    uint4 zz; zz.x = zz.y = zz.z = zz.w = 0u;
    uint4* h0q = (uint4*)(smem + SM_H0);
    uint4* h1q = (uint4*)(smem + SM_H1);
    for (int i = tid; i < (kRowsPerBlock * kH * 2) / 16; i += 256) { h0q[i] = zz; h1q[i] = zz; }
    cp16(smem + SM_WIH0, wsb + WS_WIH0, kG4 * kH * 2, tid, 256);
    cp16(smem + SM_WHH0, wsb + WS_WHH0, kG4 * kH * 2, tid, 256);
    cp16(smem + SM_WIH1, wsb + WS_WIH1, kG4 * kH * 2, tid, 256);
    cp16(smem + SM_WHH1, wsb + WS_WHH1, kG4 * kH * 2, tid, 256);
    cp16(smem + SM_WD1, wsb + WS_WD1, kDecMid * kH * 2, tid, 256);
    cp16(smem + SM_WD2, wsb + WS_WD2, kNN * kDecMid * 2, tid, 256);
  }
  __syncthreads();

  const bf16_t* s_wih0 = (const bf16_t*)(smem + SM_WIH0);
  const bf16_t* s_whh0 = (const bf16_t*)(smem + SM_WHH0);
  const bf16_t* s_wih1 = (const bf16_t*)(smem + SM_WIH1);
  const bf16_t* s_whh1 = (const bf16_t*)(smem + SM_WHH1);
  const bf16_t* s_wd1 = (const bf16_t*)(smem + SM_WD1);
  const bf16_t* s_wd2 = (const bf16_t*)(smem + SM_WD2);

  // ---- hoist loop-invariant layer-0 input term: gx0 = x@Wih0.T + bg0 ----
  v8f gx0[4][4];  // [quadrant q][pair p] -> gate tile q*4+p; 128 persistent VGPRs
#pragma unroll
  for (int q = 0; q < 4; ++q) {
#pragma unroll
    for (int p = 0; p < 4; ++p) {
      const int tn = q * 4 + p;
      v8f a = vzero8();
      a = wmma_bf16(xa0, lds_frag(s_wih0, tn * 16, 0, kH, lane), a);
      a = wmma_bf16(xa1, lds_frag(s_wih0, tn * 16, 32, kH, lane), a);
      const float bb = s_bias[BO_G0 + tn * 16 + ln];
#pragma unroll
      for (int r = 0; r < 8; ++r) a[r] += bb;
      gx0[q][p] = a;
    }
  }

  v8f c0[4], c1[4];
#pragma unroll
  for (int p = 0; p < 4; ++p) { c0[p] = vzero8(); c1[p] = vzero8(); }

  // ---- 64 recurrent steps: no barriers (each wave owns its 16 LDS rows) ----
  for (int t = 0; t < kSeq; ++t) {
    lstm_layer0(gx0, s_h0, s_whh0, c0, wloc, lane);
    v16bf h0a0 = lds_frag(s_h0, wloc, 0, kH, lane);
    v16bf h0a1 = lds_frag(s_h0, wloc, 32, kH, lane);
    lstm_layer1(h0a0, h0a1, s_h1, s_wih1, s_whh1, s_bias + BO_G1, c1, wloc, lane);

    // decoder layer 1: a1 = relu(h1 @ Wd1.T + bd1)
    v16bf hb0 = lds_frag(s_h1, wloc, 0, kH, lane);
    v16bf hb1 = lds_frag(s_h1, wloc, 32, kH, lane);
#pragma unroll
    for (int tn = 0; tn < 6; ++tn) {  // DEC_MID = 96 -> 6 N-tiles
      v8f a = vzero8();
      a = wmma_bf16(hb0, lds_frag(s_wd1, tn * 16, 0, kH, lane), a);
      a = wmma_bf16(hb1, lds_frag(s_wd1, tn * 16, 32, kH, lane), a);
      const float bb = s_bias[BO_D1 + tn * 16 + ln];
      v8f rl;
#pragma unroll
      for (int r = 0; r < 8; ++r) rl[r] = fmaxf(a[r] + bb, 0.0f);
      storeC(s_a1, wloc, tn * 16, kDecMid, lane, rl);
    }
    // decoder layer 2: logits = a1 @ Wd2.T + bd2
    v16bf aa0 = lds_frag(s_a1, wloc, 0, kDecMid, lane);
    v16bf aa1 = lds_frag(s_a1, wloc, 32, kDecMid, lane);
    v16bf aa2 = lds_frag(s_a1, wloc, 64, kDecMid, lane);
    v8f lg[8];
#pragma unroll
    for (int tn = 0; tn < 8; ++tn) {  // NN = 128 -> 8 N-tiles
      v8f a = vzero8();
      a = wmma_bf16(aa0, lds_frag(s_wd2, tn * 16, 0, kDecMid, lane), a);
      a = wmma_bf16(aa1, lds_frag(s_wd2, tn * 16, 32, kDecMid, lane), a);
      a = wmma_bf16(aa2, lds_frag(s_wd2, tn * 16, 64, kDecMid, lane), a);
      const float bb = s_bias[BO_D2 + tn * 16 + ln];
#pragma unroll
      for (int r = 0; r < 8; ++r) a[r] += bb;
      lg[tn] = a;
    }

    // softmax + Gumbel-max sample per row; rows live in 16-lane halves
#pragma unroll
    for (int r = 0; r < 8; ++r) {
      const int rowg = (int)grow + mh + r;
      float vals[8];
      float mx = -3.4e38f;
#pragma unroll
      for (int tn = 0; tn < 8; ++tn) { vals[tn] = lg[tn][r]; mx = fmaxf(mx, vals[tn]); }
#pragma unroll
      for (int s = 1; s < 16; s <<= 1) mx = fmaxf(mx, __shfl_xor(mx, s, 32));
      float es[8], sum = 0.0f;
#pragma unroll
      for (int tn = 0; tn < 8; ++tn) { es[tn] = __expf(vals[tn] - mx); sum += es[tn]; }
#pragma unroll
      for (int s = 1; s < 16; s <<= 1) sum += __shfl_xor(sum, s, 32);
      const float inv = 1.0f / sum;
      const long ob = (long)rowg * (kSeq * kNN) + (long)t * kNN + ln;
#pragma unroll
      for (int tn = 0; tn < 8; ++tn)  // streaming, write-once: non-temporal b32
        __builtin_nontemporal_store(es[tn] * inv, out_w + ob + tn * 16);

      float best = -3.4e38f; int bidx = 0;
#pragma unroll
      for (int tn = 0; tn < 8; ++tn) {
        const int n = tn * 16 + ln;
        const unsigned hs = wang(((unsigned)rowg * 0x9E3779B9u) ^
                                 ((unsigned)t * 0x85EBCA6Bu) ^
                                 ((unsigned)n * 0xC2B2AE35u));
        const float u = (float)(hs >> 8) * (1.0f / 16777216.0f) + 1e-12f;
        const float gb = -__logf(-__logf(u));
        const float cd = (vals[tn] - mx) + gb;
        if (cd > best) { best = cd; bidx = n; }
      }
#pragma unroll
      for (int s = 1; s < 16; s <<= 1) {
        const float ov = __shfl_xor(best, s, 32);
        const int oi = __shfl_xor(bidx, s, 32);
        if (ov > best) { best = ov; bidx = oi; }
      }
      if (ln == 0)
        __builtin_nontemporal_store((float)bidx, out_s + (long)rowg * kSeq + t);
    }
  }
}

extern "C" void kernel_launch(void* const* d_in, const int* in_sizes, int n_in,
                              void* d_out, int out_size, void* d_ws, size_t ws_size,
                              hipStream_t stream) {
  const float* z = (const float*)d_in[0];
  // d_in[1] score_tensor, d_in[2] train: unused by the forward pass
  const float* we1 = (const float*)d_in[3];
  const float* be1 = (const float*)d_in[4];
  const float* we2 = (const float*)d_in[5];
  const float* be2 = (const float*)d_in[6];
  const float* wih0 = (const float*)d_in[7];
  const float* whh0 = (const float*)d_in[8];
  const float* bih0 = (const float*)d_in[9];
  const float* bhh0 = (const float*)d_in[10];
  const float* wih1 = (const float*)d_in[11];
  const float* whh1 = (const float*)d_in[12];
  const float* bih1 = (const float*)d_in[13];
  const float* bhh1 = (const float*)d_in[14];
  const float* wd1 = (const float*)d_in[15];
  const float* bd1 = (const float*)d_in[16];
  const float* wd2 = (const float*)d_in[17];
  const float* bd2 = (const float*)d_in[18];

  bf16_t* wsb = (bf16_t*)d_ws;
  float* bg = (float*)((char*)d_ws + (size_t)WS_BF16_TOTAL * 2);
  float* out_w = (float*)d_out;
  float* out_s = out_w + (long)kB * kSeq * kNN;

  bar_decoder_prep<<<(kEncMid * kZD + 255) / 256, 256, 0, stream>>>(
      we1, we2, wih0, whh0, wih1, whh1, wd1, wd2, bih0, bhh0, bih1, bhh1, wsb, bg);
  bar_decoder_main<<<kBlocks, 256, SM_TOTAL, stream>>>(
      z, be1, be2, bd1, bd2, wsb, bg, out_w, out_s);
}
